// Deformable_KPN_11003706212425
// MI455X (gfx1250) — matrix operand order repaired and checked by reading, not compiled, not run on hardware
//
#include <hip/hip_runtime.h>
#include <hip/hip_bf16.h>
#include <stdint.h>

typedef __attribute__((ext_vector_type(16))) _Float16 v16h;
typedef __attribute__((ext_vector_type(8)))  _Float16 v8h;
typedef __attribute__((ext_vector_type(8)))  float    v8f;

#define HWD   65536   // H*W
#define WD    256
#define HD    256
#define BN    4
#define LSTRIDE 40    // f16 stride of LDS pixel rows: 80B = 20 banks (conflict-free for n=0..15), 16B aligned

// ---------------------------------------------------------------------------
// Pack conv weights (OIHW f32) into WMMA A-fragment tiles with TAP-MAJOR K:
//   k = tap*Cpad + cin   (Cpad = Cin rounded up to mult of 32)
// Tile (mt, kt) is 16x32 f16 = 256 dwords; dword (lane, v) holds the two f16
// lane `lane` keeps in A-VGPR `v` (contiguous k order):
//   m = 16*mt + (lane&15);  kk = (lane>>4)*16 + v*2
// ---------------------------------------------------------------------------
__global__ void pack_w_kernel(const float* __restrict__ w, uint32_t* __restrict__ pk,
                              int Cout, int Cin, int Cpad, int MT, int KT)
{
    int idx = blockIdx.x * blockDim.x + threadIdx.x;
    int total = MT * KT * 256;
    if (idx >= total) return;
    int t    = idx >> 8;
    int r    = idx & 255;
    int lane = r >> 3;
    int vv   = r & 7;
    int kt   = t / MT;
    int mt   = t - kt * MT;
    int mloc = lane & 15;
    int half = lane >> 4;
    int kk   = half * 16 + (vv << 1);
    int m    = mt * 16 + mloc;
    int k    = kt * 32 + kk;

    union { uint32_t u; _Float16 h[2]; } p;
    #pragma unroll
    for (int j = 0; j < 2; ++j) {
        int tap = (k + j) / Cpad;
        int cin = (k + j) - tap * Cpad;
        _Float16 hv = (_Float16)0.f;
        if (m < Cout && cin < Cin && tap < 9)
            hv = (_Float16)w[((size_t)m * Cin + cin) * 9 + tap];
        p.h[j] = hv;
    }
    pk[idx] = p.u;
}

// ---------------------------------------------------------------------------
// Build NHWC f16 inputs: net32 [B,HW,32] (ch 15..31 zero) and the first 15
// channels of the wc1 concat buffer wfeed [B,HW,224].
// ---------------------------------------------------------------------------
__global__ void prep_input_kernel(const float* __restrict__ data,   // NCHW [4,15,HW]
                                  _Float16* __restrict__ net32,
                                  _Float16* __restrict__ wfeed)
{
    long i = (long)blockIdx.x * blockDim.x + threadIdx.x;   // B*HW*32
    if (i >= (long)BN * HWD * 32) return;
    int c = (int)(i & 31);
    long bp = i >> 5;                   // b*HW + p
    int b = (int)(bp >> 16);
    int p = (int)(bp & 65535);
    _Float16 h = (_Float16)0.f;
    if (c < 15) h = (_Float16)data[((size_t)b * 15 + c) * HWD + p];
    net32[i] = h;
    if (c < 15) wfeed[bp * 224 + c] = h;
}

// ---------------------------------------------------------------------------
// Implicit-GEMM 3x3 SAME conv, NHWC, tap-major K, v_wmma_f32_16x16x32_f16.
// Block = 128 thr / 4 waves; tile = 64 pixels (4 N-subtiles) x 64 Cout.
// Per (tap, ct): stage 64px x 32ch B tile to LDS (1 coalesced v16h global
// load + b128 LDS store per thread), then each wave: 1 A load + 4 LDS
// fragment loads + 4 WMMAs.
// ---------------------------------------------------------------------------
__global__ __launch_bounds__(128)
void conv3x3_nhwc_wmma(const _Float16* __restrict__ act,  // NHWC [B,H,W,Cpad]
                       int Cpad, int CT,
                       const uint32_t* __restrict__ packA,
                       const float* __restrict__ bias, int Cout,
                       float*    __restrict__ out32, int so32,
                       _Float16* __restrict__ outA,  int sA,
                       _Float16* __restrict__ outB,  int sB, int cbB,
                       int relu)
{
    __shared__ _Float16 lb[64 * LSTRIDE];

    const int tid   = threadIdx.x;
    const int lane  = tid & 31;
    const int wv    = tid >> 5;
    const int half  = lane >> 4;
    const int n     = lane & 15;
    const int tile  = blockIdx.x;            // HW/64 tiles; 4 per image row
    const int b     = blockIdx.y;
    const int MTl   = gridDim.z * 4;
    const int mt    = blockIdx.z * 4 + wv;
    const int y     = tile >> 2;
    const int xBase = (tile & 3) << 6;

    const int lpx = tid >> 1;                // loader pixel 0..63
    const int lh  = tid & 1;                 // loader half-slice (16 ch)

    v8f acc0 = {}, acc1 = {}, acc2 = {}, acc3 = {};

    for (int tap = 0; tap < 9; ++tap) {
        const int rr  = tap / 3 - 1;
        const int ss  = tap - (tap / 3) * 3 - 1;
        const int yy  = y + rr;
        const int xxl = xBase + lpx + ss;
        const bool lok = ((unsigned)yy < (unsigned)HD) && ((unsigned)xxl < (unsigned)WD);
        const size_t rowbase = ((size_t)b * HWD + (size_t)yy * WD) * Cpad;

        for (int ct = 0; ct < CT; ++ct) {
            __syncthreads();
            v16h bv = {};
            if (lok)
                bv = *(const v16h*)&act[rowbase + (size_t)xxl * Cpad + ct * 32 + lh * 16];
            *(v16h*)&lb[lpx * LSTRIDE + lh * 16] = bv;
            __syncthreads();

            const int kt = tap * CT + ct;
            v16h af = *(const v16h*)(packA + (((size_t)kt * MTl + mt) << 8) + (lane << 3));
            if (kt + 1 < 9 * CT)
                __builtin_prefetch(packA + (((size_t)(kt + 1) * MTl + mt) << 8) + (lane << 3), 0, 0);

            v16h b0 = *(const v16h*)&lb[( 0 + n) * LSTRIDE + half * 16];
            v16h b1 = *(const v16h*)&lb[(16 + n) * LSTRIDE + half * 16];
            v16h b2 = *(const v16h*)&lb[(32 + n) * LSTRIDE + half * 16];
            v16h b3 = *(const v16h*)&lb[(48 + n) * LSTRIDE + half * 16];

            acc0 = __builtin_amdgcn_wmma_f32_16x16x32_f16(false, af, false, b0, (short)0, acc0, false, false);
            acc1 = __builtin_amdgcn_wmma_f32_16x16x32_f16(false, af, false, b1, (short)0, acc1, false, false);
            acc2 = __builtin_amdgcn_wmma_f32_16x16x32_f16(false, af, false, b2, (short)0, acc2, false, false);
            acc3 = __builtin_amdgcn_wmma_f32_16x16x32_f16(false, af, false, b3, (short)0, acc3, false, false);
        }
    }

    // ---- epilogue: bias (+ReLU), NHWC stores (vector f16, scalar f32) ----
    const int mLo = mt * 16 + half * 8;
    float br[8];
    #pragma unroll
    for (int i = 0; i < 8; ++i) br[i] = (mLo + i < Cout) ? bias[mLo + i] : 0.f;

    const v8f* accs[4] = { &acc0, &acc1, &acc2, &acc3 };
    #pragma unroll
    for (int nsub = 0; nsub < 4; ++nsub) {
        const v8f a = *accs[nsub];
        const size_t pix = (size_t)b * HWD + (size_t)y * WD + xBase + nsub * 16 + n;
        float v[8];
        #pragma unroll
        for (int i = 0; i < 8; ++i) {
            float t = a[i] + br[i];
            if (relu) t = fmaxf(t, 0.f);
            v[i] = t;
        }
        if (out32) {
            #pragma unroll
            for (int i = 0; i < 8; ++i)
                if (mLo + i < Cout) out32[pix * so32 + mLo + i] = v[i];
        }
        if (outA && (mLo + 8 <= Cout)) {
            v8h hv;
            #pragma unroll
            for (int i = 0; i < 8; ++i) hv[i] = (_Float16)v[i];
            *(v8h*)&outA[pix * sA + mLo] = hv;   // 16B-aligned: sA,mLo mult of 8
        }
        if (outB) {
            #pragma unroll
            for (int i = 0; i < 8; ++i)
                if (mLo + i < Cout) outB[pix * sB + cbB + mLo + i] = (_Float16)v[i];
        }
    }
}

// ---------------------------------------------------------------------------
// Trilinear deformable sampling. Offsets read NHWC (3 contiguous f32/thread),
// samples written f32 NCHW (network output) + f16 into wfeed channel slice.
// ---------------------------------------------------------------------------
__global__ void trilinear_kernel(const float* __restrict__ data,   // [4,5,3,HW]
                                 const float* __restrict__ off,    // NHWC [4,HW,81]
                                 float*    __restrict__ smp32,     // [4,27,3,HW]
                                 _Float16* __restrict__ wfeed)     // [4,HW,224], ch 143+
{
    long i = (long)blockIdx.x * blockDim.x + threadIdx.x;   // B*27*HW
    if (i >= (long)BN * 27 * HWD) return;
    int p  = (int)(i % HWD);
    int s  = (int)((i / HWD) % 27);
    int b  = (int)(i / ((long)27 * HWD));
    int hh = p >> 8;
    int wp = p & 255;

    float kd = (float)(s / 9 - 1);
    float kh = (float)((s / 3) % 3 - 1);
    float kw = (float)(s % 3 - 1);

    const float* ob = off + ((size_t)b * HWD + p) * 81 + s * 3;
    float pd = 2.0f + kd + ob[0];
    float ph = (float)hh + kh + ob[1];
    float pw = (float)wp + kw + ob[2];
    pd = fminf(fmaxf(pd, 0.f), 4.f);
    ph = fminf(fmaxf(ph, 0.f), 255.f);
    pw = fminf(fmaxf(pw, 0.f), 255.f);

    float d0f = floorf(pd), h0f = floorf(ph), w0f = floorf(pw);
    float fd = pd - d0f, fh = ph - h0f, fw = pw - w0f;
    int d0 = (int)d0f, h0 = (int)h0f, w0 = (int)w0f;
    int d1 = min(d0 + 1, 4), h1 = min(h0 + 1, 255), w1 = min(w0 + 1, 255);

    #pragma unroll
    for (int c = 0; c < 3; ++c) {
        auto tap = [&](int dd, int hy, int wx) -> float {
            return data[((((size_t)b * 5 + dd) * 3 + c) * HWD) + (size_t)hy * WD + wx];
        };
        float v =
            (1.f - fd) * ((1.f - fh) * ((1.f - fw) * tap(d0, h0, w0) + fw * tap(d0, h0, w1))
                        +        fh  * ((1.f - fw) * tap(d0, h1, w0) + fw * tap(d0, h1, w1)))
          +        fd  * ((1.f - fh) * ((1.f - fw) * tap(d1, h0, w0) + fw * tap(d1, h0, w1))
                        +        fh  * ((1.f - fw) * tap(d1, h1, w0) + fw * tap(d1, h1, w1)));
        smp32[(((size_t)b * 27 + s) * 3 + c) * HWD + p] = v;
        wfeed[((size_t)b * HWD + p) * 224 + 143 + s * 3 + c] = (_Float16)v;
    }
}

// ---------------------------------------------------------------------------
// KPN reduction: res_i[b,g,c] = 3 * sum_k s[b,g*9+k,c]*w[b,g*9+k];
// out = mean over g. Weights read NHWC (27 contiguous f32/thread).
// ---------------------------------------------------------------------------
__global__ void kpn_kernel(const float* __restrict__ smp32,  // [4,27,3,HW]
                           const float* __restrict__ wt,     // NHWC [4,HW,27]
                           float* __restrict__ res,          // [4,3,3,HW]
                           float* __restrict__ outm)         // [4,3,HW]
{
    long i = (long)blockIdx.x * blockDim.x + threadIdx.x;    // B*3*HW
    if (i >= (long)BN * 3 * HWD) return;
    int p = (int)(i % HWD);
    int c = (int)((i / HWD) % 3);
    int b = (int)(i / ((long)3 * HWD));
    const float* wp = wt + ((size_t)b * HWD + p) * 27;

    float osum = 0.f;
    #pragma unroll
    for (int g = 0; g < 3; ++g) {
        float r = 0.f;
        #pragma unroll
        for (int k = 0; k < 9; ++k) {
            int s = g * 9 + k;
            r += smp32[(((size_t)b * 27 + s) * 3 + c) * HWD + p] * wp[s];
        }
        r *= 3.0f;
        res[(((size_t)b * 3 + g) * 3 + c) * HWD + p] = r;
        osum += r;
    }
    outm[((size_t)b * 3 + c) * HWD + p] = osum / 3.0f;
}

// ---------------------------------------------------------------------------
extern "C" void kernel_launch(void* const* d_in, const int* in_sizes, int n_in,
                              void* d_out, int out_size, void* d_ws, size_t ws_size,
                              hipStream_t stream)
{
    const float* data  = (const float*)d_in[0];
    const float* w_[6] = { (const float*)d_in[1], (const float*)d_in[3], (const float*)d_in[5],
                           (const float*)d_in[7], (const float*)d_in[9], (const float*)d_in[11] };
    const float* b_[6] = { (const float*)d_in[2], (const float*)d_in[4], (const float*)d_in[6],
                           (const float*)d_in[8], (const float*)d_in[10], (const float*)d_in[12] };

    // layers: enc1, enc2, off, wc1, wc2, wc3   (tap-major K = 9*Cpad)
    const int CIN [6] = { 15,  64, 128, 224, 64, 64 };
    const int CPAD[6] = { 32,  64, 128, 224, 64, 64 };
    const int CT  [6] = { 1,   2,  4,   7,   2,  2  };   // Cpad/32
    const int COUT[6] = { 64, 128, 81,  64,  64, 27 };
    const int MT  [6] = { 4,   8,  8,   4,   4,  4  };   // ceil(Cout/16) padded to mult of 4

    // ---- workspace ----
    size_t wo = 0;
    auto wsalloc = [&](size_t bytes) -> void* {
        void* p = (char*)d_ws + wo;
        wo += (bytes + 255) & ~(size_t)255;
        return p;
    };
    _Float16* net32 = (_Float16*)wsalloc((size_t)BN * HWD * 32  * 2);
    _Float16* wfeed = (_Float16*)wsalloc((size_t)BN * HWD * 224 * 2);
    _Float16* f1    = (_Float16*)wsalloc((size_t)BN * HWD * 64  * 2);
    _Float16* feat  = (_Float16*)wsalloc((size_t)BN * HWD * 128 * 2);
    float*    offs  = (float*)   wsalloc((size_t)BN * HWD * 81  * 4);
    _Float16* h1    = (_Float16*)wsalloc((size_t)BN * HWD * 64  * 2);
    _Float16* h2    = (_Float16*)wsalloc((size_t)BN * HWD * 64  * 2);
    float*    wt    = (float*)   wsalloc((size_t)BN * HWD * 27  * 4);
    uint32_t* pkA[6];
    for (int l = 0; l < 6; ++l)
        pkA[l] = (uint32_t*)wsalloc((size_t)MT[l] * (9 * CT[l]) * 256 * 4);

    // ---- output regions (return order: res_i, out, samples) ----
    float* outp  = (float*)d_out;
    float* res   = outp;
    float* outm  = outp + 2359296;
    float* smp32 = outp + 3145728;

    // ---- 1) inputs -> NHWC f16; weights -> WMMA A tiles (tap-major) ----
    {
        long n = (long)BN * HWD * 32;
        prep_input_kernel<<<(unsigned)((n + 255) / 256), 256, 0, stream>>>(data, net32, wfeed);
    }
    for (int l = 0; l < 6; ++l) {
        int total = MT[l] * (9 * CT[l]) * 256;
        pack_w_kernel<<<(total + 255) / 256, 256, 0, stream>>>(w_[l], pkA[l],
                                                               COUT[l], CIN[l], CPAD[l],
                                                               MT[l], 9 * CT[l]);
    }

    dim3 cg(HWD / 64, BN, 1);
    // ---- 2) enc1: 15(->32) -> 64, ReLU ----
    cg.z = MT[0] / 4;
    conv3x3_nhwc_wmma<<<cg, 128, 0, stream>>>(net32, CPAD[0], CT[0], pkA[0], b_[0], COUT[0],
                                              nullptr, 0, f1, 64, nullptr, 0, 0, 1);
    // ---- 3) enc2: 64 -> 128, ReLU (dual-write: feat + wfeed[15..142]) ----
    cg.z = MT[1] / 4;
    conv3x3_nhwc_wmma<<<cg, 128, 0, stream>>>(f1, CPAD[1], CT[1], pkA[1], b_[1], COUT[1],
                                              nullptr, 0, feat, 128, wfeed, 224, 15, 1);
    // ---- 4) offsets: 128 -> 81 (f32 NHWC) ----
    cg.z = MT[2] / 4;
    conv3x3_nhwc_wmma<<<cg, 128, 0, stream>>>(feat, CPAD[2], CT[2], pkA[2], b_[2], COUT[2],
                                              offs, 81, nullptr, 0, nullptr, 0, 0, 0);
    // ---- 5) trilinear sampling -> samples f32 (out) + wfeed[143..223] ----
    {
        long n = (long)BN * 27 * HWD;
        trilinear_kernel<<<(unsigned)((n + 255) / 256), 256, 0, stream>>>(data, offs, smp32, wfeed);
    }
    // ---- 6) wc1: 224 -> 64, ReLU ----
    cg.z = MT[3] / 4;
    conv3x3_nhwc_wmma<<<cg, 128, 0, stream>>>(wfeed, CPAD[3], CT[3], pkA[3], b_[3], COUT[3],
                                              nullptr, 0, h1, 64, nullptr, 0, 0, 1);
    // ---- 7) wc2: 64 -> 64, ReLU ----
    cg.z = MT[4] / 4;
    conv3x3_nhwc_wmma<<<cg, 128, 0, stream>>>(h1, CPAD[4], CT[4], pkA[4], b_[4], COUT[4],
                                              nullptr, 0, h2, 64, nullptr, 0, 0, 1);
    // ---- 8) wc3: 64 -> 27 (f32 NHWC) ----
    cg.z = MT[5] / 4;
    conv3x3_nhwc_wmma<<<cg, 128, 0, stream>>>(h2, CPAD[5], CT[5], pkA[5], b_[5], COUT[5],
                                              wt, 27, nullptr, 0, nullptr, 0, 0, 0);
    // ---- 9) KPN weighted sum ----
    {
        long n = (long)BN * 3 * HWD;
        kpn_kernel<<<(unsigned)((n + 255) / 256), 256, 0, stream>>>(smp32, wt, res, outm);
    }
}
